// LAttnDecoderRNN_45037027066093
// MI455X (gfx1250) — matrix operand order repaired and checked by reading, not compiled, not run on hardware
//
#include <hip/hip_runtime.h>
#include <hip/hip_bf16.h>

// ---------------------------------------------------------------------------
// LAttnDecoderRNN single step on MI455X (gfx1250, wave32, WMMA).
// HBM-bound (W_o = 206MB dominates; floor ~15us @ 23.3 TB/s). Strategy:
//  - every big operand streams exactly once (each wave owns a 16-wide N tile
//    and computes the full M=64 strip),
//  - all GEMMs use v_wmma_f32_16x16x32_f16 with ISA-exact fragment layouts,
//  - A-matrix of the dominant logit GEMM is pre-packed to f16 fragment layout
//    (kills 32 v_cvt per K-step in 3142 waves),
//  - W_o loads are non-temporal (TH=NT) + software prefetch so the once-only
//    stream does not evict L2-resident activations.
// ---------------------------------------------------------------------------

typedef __attribute__((ext_vector_type(16))) _Float16 v16h;
typedef __attribute__((ext_vector_type(8)))  float    v8f;
typedef __attribute__((ext_vector_type(2)))  float    v2f;   // real vector for NT loads

#define HDIM 1024
#define VDIM 50257
#define BDIM 64
#define TDIM 512
#define H3   3072
#define H2   2048

// ---- Fragment loaders (ISA 7.12.2 layouts, wave32) ------------------------

// A: 16x32 f16 (MxK). lanes 0-15: M=0..15; lane group g=lane>>4 selects K half.
// VGPR v<4:  K = kBase + g*8 + 2v ; VGPR v>=4: K = kBase + 16 + g*8 + 2(v-4)
__device__ inline int a_frag_k(int kBase, int g, int v) {
  return (v < 4) ? (kBase + g * 8 + 2 * v) : (kBase + 16 + g * 8 + 2 * (v - 4));
}

__device__ inline v16h load_a_frag(const float* __restrict__ A, int lda,
                                   int mBase, int kBase, int lane) {
  const int m = mBase + (lane & 15);
  const int g = lane >> 4;
  const float* row = A + (size_t)m * lda;
  v16h a;
#pragma unroll
  for (int v = 0; v < 8; ++v) {
    const v2f f = *reinterpret_cast<const v2f*>(row + a_frag_k(kBase, g, v));
    a[2 * v]     = (_Float16)f.x;
    a[2 * v + 1] = (_Float16)f.y;
  }
  return a;
}

// A loader for concat([h, ctx]) along K (K=2048, first half h, second ctx).
__device__ inline v16h load_a_frag_cat(const float* __restrict__ h,
                                       const float* __restrict__ ctx,
                                       int mBase, int kBase, int lane) {
  const int m = mBase + (lane & 15);
  const int g = lane >> 4;
  v16h a;
#pragma unroll
  for (int v = 0; v < 8; ++v) {
    const int k = a_frag_k(kBase, g, v);
    const float* src = (k < HDIM) ? (h + (size_t)m * HDIM + k)
                                  : (ctx + (size_t)m * HDIM + (k - HDIM));
    const v2f f = *reinterpret_cast<const v2f*>(src);
    a[2 * v]     = (_Float16)f.x;
    a[2 * v + 1] = (_Float16)f.y;
  }
  return a;
}

// B: 32x16 f16 (KxN) from row-major W[N][K] (B = W^T, K contiguous).
// K order matches A (a_frag_k): g selects K 8..15/24..31 half; lane&15 = N.
__device__ inline v16h load_b_frag_nk(const float* __restrict__ W, int ldw,
                                      int nBase, int kBase, int lane,
                                      int nMax) {
  int n = nBase + (lane & 15);
  if (n > nMax) n = nMax;              // clamp for ragged last tile (N=VDIM)
  const int g = lane >> 4;
  const float* row = W + (size_t)n * ldw;
  v16h b;
#pragma unroll
  for (int v = 0; v < 8; ++v) {
    const v2f f = *reinterpret_cast<const v2f*>(row + a_frag_k(kBase, g, v));
    b[2 * v]     = (_Float16)f.x;
    b[2 * v + 1] = (_Float16)f.y;
  }
  return b;
}

// Same, but non-temporal (TH=NT): used for the once-only 206MB W_o stream so
// it does not displace L2-resident activations.
__device__ inline v16h load_b_frag_nk_nt(const float* __restrict__ W, int ldw,
                                         int nBase, int kBase, int lane,
                                         int nMax) {
  int n = nBase + (lane & 15);
  if (n > nMax) n = nMax;
  const int g = lane >> 4;
  const float* row = W + (size_t)n * ldw;
  v16h b;
#pragma unroll
  for (int v = 0; v < 8; ++v) {
    const v2f f = __builtin_nontemporal_load(
        reinterpret_cast<const v2f*>(row + a_frag_k(kBase, g, v)));
    b[2 * v]     = (_Float16)f.x;
    b[2 * v + 1] = (_Float16)f.y;
  }
  return b;
}

// B: 32x16 f16 from K-major storage W[K][N] (used for W_a: B[k][n]=W_a[k][n]).
__device__ inline v16h load_b_frag_kn(const float* __restrict__ W, int ldw,
                                      int nBase, int kBase, int lane) {
  const int n = nBase + (lane & 15);
  const int g = lane >> 4;
  v16h b;
#pragma unroll
  for (int v = 0; v < 8; ++v) {
    const int k = a_frag_k(kBase, g, v);
    b[2 * v]     = (_Float16)W[(size_t)k * ldw + n];
    b[2 * v + 1] = (_Float16)W[(size_t)(k + 1) * ldw + n];
  }
  return b;
}

#define WMMA_F16(a, b, c) \
  __builtin_amdgcn_wmma_f32_16x16x32_f16(false, (a), false, (b), (short)0, (c), false, false)

// ---- Kernel 0: embedding gather x = emb[input_seq] ------------------------
__global__ void gather_x_kernel(const int* __restrict__ idx,
                                const float* __restrict__ emb,
                                float* __restrict__ x) {
  const int b = blockIdx.x;
  const int row = idx[b];
  for (int h = threadIdx.x; h < HDIM; h += blockDim.x)
    x[(size_t)b * HDIM + h] = emb[(size_t)row * HDIM + h];
}

// ---- Kernel 1: gi = x@W_ih^T + b_ih ; gh = h0@W_hh^T + b_hh (wmma) --------
__global__ void gru_gemm_kernel(const float* __restrict__ x,
                                const float* __restrict__ h0,
                                const float* __restrict__ W_ih,
                                const float* __restrict__ W_hh,
                                const float* __restrict__ b_ih,
                                const float* __restrict__ b_hh,
                                float* __restrict__ gi,
                                float* __restrict__ gh) {
  const int wave = (blockIdx.x * blockDim.x + threadIdx.x) >> 5;
  const int lane = threadIdx.x & 31;
  const int which = wave / (H3 / 16);      // 0 -> gi, 1 -> gh
  const int nTile = wave % (H3 / 16);
  const float* A    = which ? h0   : x;
  const float* W    = which ? W_hh : W_ih;
  const float* bias = which ? b_hh : b_ih;
  float* out        = which ? gh   : gi;
  const int nBase = nTile * 16;

  v8f c0 = {}, c1 = {}, c2 = {}, c3 = {};
  for (int kBase = 0; kBase < HDIM; kBase += 32) {
    const v16h bf = load_b_frag_nk(W, HDIM, nBase, kBase, lane, H3 - 1);
    const v16h a0 = load_a_frag(A, HDIM, 0,  kBase, lane);
    const v16h a1 = load_a_frag(A, HDIM, 16, kBase, lane);
    const v16h a2 = load_a_frag(A, HDIM, 32, kBase, lane);
    const v16h a3 = load_a_frag(A, HDIM, 48, kBase, lane);
    c0 = WMMA_F16(a0, bf, c0);
    c1 = WMMA_F16(a1, bf, c1);
    c2 = WMMA_F16(a2, bf, c2);
    c3 = WMMA_F16(a3, bf, c3);
  }
  const int g = lane >> 4;
  const int n = nBase + (lane & 15);
  const float bv = bias[n];
#pragma unroll
  for (int r = 0; r < 8; ++r) {
    const int m = r + 8 * g;
    out[(size_t)(m)      * H3 + n] = c0[r] + bv;
    out[(size_t)(m + 16) * H3 + n] = c1[r] + bv;
    out[(size_t)(m + 32) * H3 + n] = c2[r] + bv;
    out[(size_t)(m + 48) * H3 + n] = c3[r] + bv;
  }
}

// ---- Kernel 2: GRU gate combine, torch order (r,z,n) ----------------------
__global__ void gru_combine_kernel(const float* __restrict__ gi,
                                   const float* __restrict__ gh,
                                   const float* __restrict__ h0,
                                   float* __restrict__ hOut) {
  const int idx = blockIdx.x * blockDim.x + threadIdx.x;
  if (idx >= BDIM * HDIM) return;
  const int b = idx / HDIM, hc = idx % HDIM;
  const float* gib = gi + (size_t)b * H3;
  const float* ghb = gh + (size_t)b * H3;
  const float r = 1.f / (1.f + __expf(-(gib[hc] + ghb[hc])));
  const float z = 1.f / (1.f + __expf(-(gib[HDIM + hc] + ghb[HDIM + hc])));
  const float n = tanhf(gib[2 * HDIM + hc] + r * ghb[2 * HDIM + hc]);
  hOut[idx] = (1.f - z) * n + z * h0[idx];
}

// ---- Kernel 3: hW = h @ W_a (K-major B), wmma. b_a dropped (softmax-inv.) -
__global__ void hw_gemm_kernel(const float* __restrict__ hmat,
                               const float* __restrict__ W_a,
                               float* __restrict__ hW) {
  const int wave = (blockIdx.x * blockDim.x + threadIdx.x) >> 5;  // 64 waves
  const int lane = threadIdx.x & 31;
  const int nBase = wave * 16;

  v8f c0 = {}, c1 = {}, c2 = {}, c3 = {};
  for (int kBase = 0; kBase < HDIM; kBase += 32) {
    const v16h bf = load_b_frag_kn(W_a, HDIM, nBase, kBase, lane);
    const v16h a0 = load_a_frag(hmat, HDIM, 0,  kBase, lane);
    const v16h a1 = load_a_frag(hmat, HDIM, 16, kBase, lane);
    const v16h a2 = load_a_frag(hmat, HDIM, 32, kBase, lane);
    const v16h a3 = load_a_frag(hmat, HDIM, 48, kBase, lane);
    c0 = WMMA_F16(a0, bf, c0);
    c1 = WMMA_F16(a1, bf, c1);
    c2 = WMMA_F16(a2, bf, c2);
    c3 = WMMA_F16(a3, bf, c3);
  }
  const int g = lane >> 4;
  const int n = nBase + (lane & 15);
#pragma unroll
  for (int r = 0; r < 8; ++r) {
    const int m = r + 8 * g;
    hW[(size_t)(m)      * HDIM + n] = c0[r];
    hW[(size_t)(m + 16) * HDIM + n] = c1[r];
    hW[(size_t)(m + 32) * HDIM + n] = c2[r];
    hW[(size_t)(m + 48) * HDIM + n] = c3[r];
  }
}

// ---- Kernel 4: fused scores -> softmax -> context, one block per b --------
__global__ void attn_kernel(const float* __restrict__ hW,
                            const float* __restrict__ enc,
                            float* __restrict__ ctx) {
  __shared__ float sc[TDIM];
  __shared__ float red[8];
  const int b = blockIdx.x;
  const int tid = threadIdx.x;
  const int lane = tid & 31, wave = tid >> 5;   // 8 waves of 32
  const float* encB = enc + (size_t)b * TDIM * HDIM;
  const float* hWb  = hW + (size_t)b * HDIM;

  // scores[t] = dot(hW[b], enc[b,t,:]) — lane-parallel over H, coalesced
  for (int t = wave; t < TDIM; t += 8) {
    const float* row = encB + (size_t)t * HDIM;
    float acc = 0.f;
    for (int i = lane; i < HDIM; i += 32) acc += hWb[i] * row[i];
#pragma unroll
    for (int off = 16; off > 0; off >>= 1) acc += __shfl_xor(acc, off, 32);
    if (lane == 0) sc[t] = acc;
  }
  __syncthreads();

  // softmax: block max
  float m = -3.402823466e38f;
  for (int t = tid; t < TDIM; t += 256) m = fmaxf(m, sc[t]);
#pragma unroll
  for (int off = 16; off > 0; off >>= 1) m = fmaxf(m, __shfl_xor(m, off, 32));
  if (lane == 0) red[wave] = m;
  __syncthreads();
  if (tid == 0) {
    float mm = red[0];
    for (int i = 1; i < 8; ++i) mm = fmaxf(mm, red[i]);
    red[0] = mm;
  }
  __syncthreads();
  const float bm = red[0];
  __syncthreads();

  // exp + block sum
  float s = 0.f;
  for (int t = tid; t < TDIM; t += 256) {
    const float e = __expf(sc[t] - bm);
    sc[t] = e;
    s += e;
  }
#pragma unroll
  for (int off = 16; off > 0; off >>= 1) s += __shfl_xor(s, off, 32);
  if (lane == 0) red[wave] = s;
  __syncthreads();
  if (tid == 0) {
    float ss = 0.f;
    for (int i = 0; i < 8; ++i) ss += red[i];
    red[0] = ss;
  }
  __syncthreads();
  const float inv = 1.f / red[0];
  for (int t = tid; t < TDIM; t += 256) sc[t] *= inv;
  __syncthreads();

  // context[b,h] = sum_t w[t] * enc[b,t,h] — coalesced across tid
#pragma unroll
  for (int j = 0; j < 4; ++j) {
    const int hcol = tid + 256 * j;
    float acc = 0.f;
    for (int t = 0; t < TDIM; ++t) acc += sc[t] * encB[(size_t)t * HDIM + hcol];
    ctx[(size_t)b * HDIM + hcol] = acc;
  }
}

// ---- Kernel 5: catout = tanh([h,ctx] @ W_c^T + b_c), K=2048, wmma ---------
__global__ void cat_gemm_kernel(const float* __restrict__ hmat,
                                const float* __restrict__ ctx,
                                const float* __restrict__ W_c,
                                const float* __restrict__ b_c,
                                float* __restrict__ catout) {
  const int wave = (blockIdx.x * blockDim.x + threadIdx.x) >> 5;  // 64 waves
  const int lane = threadIdx.x & 31;
  const int nBase = wave * 16;

  v8f c0 = {}, c1 = {}, c2 = {}, c3 = {};
  for (int kBase = 0; kBase < H2; kBase += 32) {
    const v16h bf = load_b_frag_nk(W_c, H2, nBase, kBase, lane, HDIM - 1);
    const v16h a0 = load_a_frag_cat(hmat, ctx, 0,  kBase, lane);
    const v16h a1 = load_a_frag_cat(hmat, ctx, 16, kBase, lane);
    const v16h a2 = load_a_frag_cat(hmat, ctx, 32, kBase, lane);
    const v16h a3 = load_a_frag_cat(hmat, ctx, 48, kBase, lane);
    c0 = WMMA_F16(a0, bf, c0);
    c1 = WMMA_F16(a1, bf, c1);
    c2 = WMMA_F16(a2, bf, c2);
    c3 = WMMA_F16(a3, bf, c3);
  }
  const int g = lane >> 4;
  const int n = nBase + (lane & 15);
  const float bv = b_c[n];
#pragma unroll
  for (int r = 0; r < 8; ++r) {
    const int m = r + 8 * g;
    catout[(size_t)(m)      * HDIM + n] = tanhf(c0[r] + bv);
    catout[(size_t)(m + 16) * HDIM + n] = tanhf(c1[r] + bv);
    catout[(size_t)(m + 32) * HDIM + n] = tanhf(c2[r] + bv);
    catout[(size_t)(m + 48) * HDIM + n] = tanhf(c3[r] + bv);
  }
}

// ---- Kernel 5b: pre-pack catout (64x1024 f32) into f16 A-fragment layout --
// aPack[((mTile*32 + kTile)*32 + lane)] = v16h fragment. 128KB total, so the
// 3142 waves of the logit GEMM load A with zero conversions.
__global__ void pack_a_kernel(const float* __restrict__ catout,
                              _Float16* __restrict__ aPack) {
  const int gid = blockIdx.x * blockDim.x + threadIdx.x;   // 4*32*32 = 4096
  if (gid >= 4 * 32 * 32) return;
  const int mTile = gid >> 10;
  const int kTile = (gid >> 5) & 31;
  const int lane  = gid & 31;
  const int m = mTile * 16 + (lane & 15);
  const int g = lane >> 4;
  const float* row = catout + (size_t)m * HDIM;
  _Float16* dst = aPack + (size_t)gid * 16;
#pragma unroll
  for (int v = 0; v < 8; ++v) {
    const v2f f = *reinterpret_cast<const v2f*>(row + a_frag_k(kTile * 32, g, v));
    dst[2 * v]     = (_Float16)f.x;
    dst[2 * v + 1] = (_Float16)f.y;
  }
}

// ---- Kernel 6: logits = aPack @ W_o^T + b_o (N=50257), wmma ---------------
// W_o streams exactly once, non-temporal, with software prefetch ahead.
#define NTILES_OUT ((VDIM + 15) / 16)   // 3142
__global__ void out_gemm_kernel(const _Float16* __restrict__ aPack,
                                const float* __restrict__ W_o,
                                const float* __restrict__ b_o,
                                float* __restrict__ out) {
  const int wave = (blockIdx.x * blockDim.x + threadIdx.x) >> 5;
  if (wave >= NTILES_OUT) return;          // wave-uniform (EXEC stays full)
  const int lane = threadIdx.x & 31;
  const int nBase = wave * 16;
  int nClamped = nBase + (lane & 15);
  if (nClamped > VDIM - 1) nClamped = VDIM - 1;
  const float* wRow = W_o + (size_t)nClamped * HDIM;

  v8f c0 = {}, c1 = {}, c2 = {}, c3 = {};
  for (int kTile = 0; kTile < 32; ++kTile) {
    const int kBase = kTile * 32;
    // stream this lane's W_o row ahead (speculative, in-row)
    if (kBase + 288 < HDIM) __builtin_prefetch(wRow + kBase + 288, 0, 0);
    const v16h bf = load_b_frag_nk_nt(W_o, HDIM, nBase, kBase, lane, VDIM - 1);
    const v16h* aP = reinterpret_cast<const v16h*>(aPack) + (size_t)kTile * 32 + lane;
    const v16h a0 = aP[0 * 32 * 32];
    const v16h a1 = aP[1 * 32 * 32];
    const v16h a2 = aP[2 * 32 * 32];
    const v16h a3 = aP[3 * 32 * 32];
    c0 = WMMA_F16(a0, bf, c0);
    c1 = WMMA_F16(a1, bf, c1);
    c2 = WMMA_F16(a2, bf, c2);
    c3 = WMMA_F16(a3, bf, c3);
  }
  const int g = lane >> 4;
  const int n = nBase + (lane & 15);
  if (n < VDIM) {
    const float bv = b_o[n];
#pragma unroll
    for (int r = 0; r < 8; ++r) {
      const int m = r + 8 * g;
      out[(size_t)(m)      * VDIM + n] = c0[r] + bv;
      out[(size_t)(m + 16) * VDIM + n] = c1[r] + bv;
      out[(size_t)(m + 32) * VDIM + n] = c2[r] + bv;
      out[(size_t)(m + 48) * VDIM + n] = c3[r] + bv;
    }
  }
}

// ---------------------------------------------------------------------------
extern "C" void kernel_launch(void* const* d_in, const int* in_sizes, int n_in,
                              void* d_out, int out_size, void* d_ws, size_t ws_size,
                              hipStream_t stream) {
  (void)in_sizes; (void)n_in; (void)out_size; (void)ws_size;
  const int*   input_seq = (const int*)  d_in[0];
  const float* last_hid  = (const float*)d_in[1];   // [1,B,H] -> [B,H]
  const float* enc       = (const float*)d_in[2];   // [B,T,H]
  const float* emb       = (const float*)d_in[3];   // [V,H]
  const float* W_ih      = (const float*)d_in[4];   // [3H,H]
  const float* W_hh      = (const float*)d_in[5];   // [3H,H]
  const float* b_ih      = (const float*)d_in[6];
  const float* b_hh      = (const float*)d_in[7];
  const float* W_a       = (const float*)d_in[8];   // [H,H]
  // d_in[9] = b_a: per-row-constant score shift -> softmax-invariant, unused.
  const float* W_c       = (const float*)d_in[10];  // [H,2H]
  const float* b_c       = (const float*)d_in[11];
  const float* W_o       = (const float*)d_in[12];  // [V,H]
  const float* b_o       = (const float*)d_in[13];

  float* out    = (float*)d_out;                 // [B,V] logits
  float* hidden = out + (size_t)BDIM * VDIM;     // [B,H] new hidden

  float* ws   = (float*)d_ws;
  float* x    = ws;                              //  64k floats
  float* gi   = ws + 65536;                      // 192k
  float* gh   = ws + 262144;                     // 192k
  float* hW   = ws + 458752;                     //  64k
  float* ctx  = ws + 524288;                     //  64k
  float* cato = ws + 589824;                     //  64k
  _Float16* aPack = (_Float16*)(ws + 655360);    //  64k halves (128KB)

  gather_x_kernel<<<BDIM, 256, 0, stream>>>(input_seq, emb, x);
  gru_gemm_kernel<<<48, 256, 0, stream>>>(x, last_hid, W_ih, W_hh, b_ih, b_hh, gi, gh);
  gru_combine_kernel<<<(BDIM * HDIM) / 256, 256, 0, stream>>>(gi, gh, last_hid, hidden);
  hw_gemm_kernel<<<8, 256, 0, stream>>>(hidden, W_a, hW);
  attn_kernel<<<BDIM, 256, 0, stream>>>(hW, enc, ctx);
  cat_gemm_kernel<<<8, 256, 0, stream>>>(hidden, ctx, W_c, b_c, cato);
  pack_a_kernel<<<16, 256, 0, stream>>>(cato, aPack);
  out_gemm_kernel<<<(NTILES_OUT + 7) / 8, 256, 0, stream>>>(aPack, W_o, b_o, out);
}
